// LiDARBEVEncoder_38903813767724
// MI455X (gfx1250) — compile-verified, matrix operand order checked by reading, not run on hardware
//
#include <hip/hip_runtime.h>
#include <hip/hip_bf16.h>
#include <cstdint>
#include <cstddef>

// ---------------- types for WMMA ----------------
typedef __attribute__((ext_vector_type(16))) _Float16 v16h;
typedef __attribute__((ext_vector_type(8)))  _Float16 v8h;
typedef __attribute__((ext_vector_type(8)))  float    v8f;

#define GXn 400
#define GYn 400
#define NSEG (GXn * GYn)

// GEMM tiling
#define BM 128
#define BN 64
#define BK 32
#define LDT 40   // padded LDS row stride (halves): 80B rows -> conflict-free frag reads

// ======================================================================
// small helper kernels
// ======================================================================
__global__ void k_bnfold(const float* __restrict__ g, const float* __restrict__ b,
                         const float* __restrict__ m, const float* __restrict__ v,
                         float* __restrict__ scale, float* __restrict__ shift, int n)
{
    int i = blockIdx.x * blockDim.x + threadIdx.x;
    if (i >= n) return;
    float s = g[i] * rsqrtf(v[i] + 1e-3f);
    scale[i] = s;
    shift[i] = b[i] - m[i] * s;
}

__global__ void k_voxelize(const float* __restrict__ pts, int n,
                           float* __restrict__ counts, float* __restrict__ sums,
                           int* __restrict__ vids)
{
    int i = blockIdx.x * blockDim.x + threadIdx.x;
    if (i >= n) return;
    float x = pts[i*5+1], y = pts[i*5+2], z = pts[i*5+3];
    int xi = (int)floorf((x + 20.0f) * 10.0f);
    xi = min(max(xi, 0), GXn - 1);
    int yi = (int)floorf((y + 20.0f) * 10.0f);
    yi = min(max(yi, 0), GYn - 1);
    int vid = xi * GYn + yi;             // batch_size == 1
    vids[i] = vid;
    atomicAdd(&counts[vid], 1.0f);
    atomicAdd(&sums[vid*3+0], x);
    atomicAdd(&sums[vid*3+1], y);
    atomicAdd(&sums[vid*3+2], z);
}

// PFN: 9->64 matmul + BN + ReLU per point, then pillar max-pool via atomicMax on
// float bits (ReLU output is >=0, pooled is zero-initialized, so bit-max == fp-max
// and empty pillars stay 0, matching the reference).
__global__ __launch_bounds__(256) void k_pfn(
    const float* __restrict__ pts, int n, const int* __restrict__ vids,
    const float* __restrict__ counts, const float* __restrict__ sums,
    const float* __restrict__ W, const float* __restrict__ scale,
    const float* __restrict__ shift, float* __restrict__ pooled)
{
    __shared__ float sW[9 * 64];
    __shared__ float sS[64], sH[64];
    for (int t = threadIdx.x; t < 9 * 64; t += blockDim.x) sW[t] = W[t];
    for (int t = threadIdx.x; t < 64; t += blockDim.x) { sS[t] = scale[t]; sH[t] = shift[t]; }
    __syncthreads();

    int i = blockIdx.x * blockDim.x + threadIdx.x;
    if (i >= n) return;
    float x = pts[i*5+1], y = pts[i*5+2], z = pts[i*5+3], r = pts[i*5+4];
    int vid = vids[i];
    float cnt = fmaxf(counts[vid], 1.0f);
    float mx = sums[vid*3+0] / cnt, my = sums[vid*3+1] / cnt, mz = sums[vid*3+2] / cnt;
    int xi = vid / GYn, yi = vid % GYn;
    float px = (float)xi * 0.1f + 0.05f - 20.0f;
    float py = (float)yi * 0.1f + 0.05f - 20.0f;
    float f[9] = { x, y, z, r, x - mx, y - my, z - mz, x - px, y - py };

    unsigned* dst = (unsigned*)(pooled + (size_t)vid * 64);
    for (int c = 0; c < 64; ++c) {
        float a = 0.f;
        #pragma unroll
        for (int j = 0; j < 9; ++j) a += f[j] * sW[j * 64 + c];
        float hv = fmaxf(a * sS[c] + sH[c], 0.0f);
        if (hv > 0.0f) atomicMax(dst + c, __float_as_uint(hv));
    }
}

__global__ void k_cvt_f32_f16(const float* __restrict__ src, _Float16* __restrict__ dst, int n)
{
    int i = blockIdx.x * blockDim.x + threadIdx.x;
    if (i < n) dst[i] = (_Float16)src[i];
}

// OIHW f32 -> [O][ky][kx][I] f16  (K = tap-major, channel-minor)
__global__ void k_repack_conv(const float* __restrict__ w, _Float16* __restrict__ wt,
                              int cout, int cin)
{
    int i = blockIdx.x * blockDim.x + threadIdx.x;
    int total = cout * cin * 9;
    if (i >= total) return;
    int t = i;
    int kx = t % 3; t /= 3;
    int ky = t % 3; t /= 3;
    int ci = t % cin; int o = t / cin;
    wt[((size_t)o * 9 + ky * 3 + kx) * cin + ci] = (_Float16)w[i];
}

// ConvTranspose (Cin,Cout,k,k) f32 -> [tap][O][Cin] f16
__global__ void k_repack_deconv(const float* __restrict__ w, _Float16* __restrict__ wt,
                                int cin, int cout, int k)
{
    int i = blockIdx.x * blockDim.x + threadIdx.x;
    int total = cin * cout * k * k;
    if (i >= total) return;
    int t = i;
    int kx = t % k; t /= k;
    int ky = t % k; t /= k;
    int o = t % cout; int ci = t / cout;
    wt[((size_t)(ky * k + kx) * cout + o) * cin + ci] = (_Float16)w[i];
}

// ======================================================================
// Implicit-GEMM 3x3 conv (pad 1) with WMMA f16 -> f32, BN+ReLU epilogue.
// K loop is tap-major (9 taps unrolled, constant ky/kx, per-tap halo check
// done once) x channel chunks of 32, with register prefetch of the next
// A/B chunk overlapping global latency with ds_load+WMMA of the current one.
// mode 0: store NHWC f16 at channel offset/stride (concat); mode 1: f32 NCHW.
// ======================================================================
__global__ __launch_bounds__(256) void k_conv3x3_wmma(
    const _Float16* __restrict__ in, int Hin, int Win, int Cin, int stride,
    const _Float16* __restrict__ wt,
    const float* __restrict__ scale, const float* __restrict__ shift,
    _Float16* __restrict__ out16, float* __restrict__ out32,
    int Hout, int Wout, int coffset, int cstride, int mode)
{
    __shared__ _Float16 sA[BM * LDT];
    __shared__ _Float16 sB[BN * LDT];

    const int M = Hout * Wout;
    const int K = 9 * Cin;
    const int m0 = blockIdx.x * BM;
    const int n0 = blockIdx.y * BN;
    const int tid = threadIdx.x;
    const int lane = tid & 31, wv = tid >> 5;

    // A staging coords: 2 threads per row, 16 halves each (Cin % 32 == 0,
    // so a 16-wide chunk never crosses a tap boundary). One-time divisions.
    const int ar = tid >> 1;
    const int ah = (tid & 1) * 16;
    const int am = m0 + ar;
    const int aoy = am / Wout, aox = am % Wout;
    const int aiy0 = aoy * stride - 1;   // + ky
    const int aix0 = aox * stride - 1;   // + kx
    const bool arow_ok = (am < M);
    uint4* const adst = (uint4*)(sA + ar * LDT + ah);

    // B staging coords (threads 0..127)
    const int bn = tid >> 1;
    const int bh = (tid & 1) * 16;
    const bool bldr = (tid < 128);
    const _Float16* const brow = wt + (size_t)(n0 + bn) * K + bh;
    uint4* const bdst = (uint4*)(sB + bn * LDT + bh);

    // fragment LDS read pointers (ISA 16-bit 16x32 lane layout)
    const int fr = lane & 15;
    const int kb = (lane >> 4) * 8;
    const _Float16* const paf = sA + (wv * 16 + fr) * LDT + kb;
    const _Float16* const pbf = sB + fr * LDT + kb;

    v8f acc[4] = {};

    #pragma unroll
    for (int tap = 0; tap < 9; ++tap) {
        const int ky = tap / 3, kx = tap % 3;          // compile-time after unroll
        const int iy = aiy0 + ky, ix = aix0 + kx;
        const bool valid = arow_ok & ((unsigned)iy < (unsigned)Hin)
                                   & ((unsigned)ix < (unsigned)Win);
        const _Float16* const asrc = in + ((size_t)iy * Win + ix) * Cin + ah;
        const _Float16* const bsrc = brow + tap * Cin;

        // prefetch first chunk of this tap
        uint4 a0 = {0,0,0,0}, a1 = {0,0,0,0};
        uint4 b0 = {0,0,0,0}, b1 = {0,0,0,0};
        if (valid) { const uint4* s = (const uint4*)asrc; a0 = s[0]; a1 = s[1]; }
        if (bldr)  { const uint4* s = (const uint4*)bsrc; b0 = s[0]; b1 = s[1]; }

        for (int c0 = 0; c0 < Cin; c0 += BK) {
            // commit staged chunk to LDS
            adst[0] = a0; adst[1] = a1;
            if (bldr) { bdst[0] = b0; bdst[1] = b1; }
            __syncthreads();

            // prefetch next chunk (overlaps with ds_load + WMMA below)
            if (c0 + BK < Cin) {
                a0 = uint4{0,0,0,0}; a1 = uint4{0,0,0,0};
                if (valid) { const uint4* s = (const uint4*)(asrc + c0 + BK); a0 = s[0]; a1 = s[1]; }
                if (bldr)  { const uint4* s = (const uint4*)(bsrc + c0 + BK); b0 = s[0]; b1 = s[1]; }
            }

            // fragments + WMMA
            v8h alo = *(const v8h*)paf;
            v8h ahi = *(const v8h*)(paf + 16);
            v16h a = __builtin_shufflevector(alo, ahi, 0,1,2,3,4,5,6,7,8,9,10,11,12,13,14,15);

            #pragma unroll
            for (int j = 0; j < 4; ++j) {
                const _Float16* pb = pbf + j * 16 * LDT;
                v8h blo = *(const v8h*)pb;
                v8h bhi = *(const v8h*)(pb + 16);
                v16h bb = __builtin_shufflevector(blo, bhi, 0,1,2,3,4,5,6,7,8,9,10,11,12,13,14,15);
                acc[j] = __builtin_amdgcn_wmma_f32_16x16x32_f16(
                    false, a, false, bb, (short)0, acc[j], false, false);
            }
            __syncthreads();
        }
    }

    // ---- epilogue: BN affine + ReLU ----
    const int mbase = wv * 16 + (lane >> 4) * 8;
    const int nf = lane & 15;
    #pragma unroll
    for (int j = 0; j < 4; ++j) {
        int n = n0 + j * 16 + nf;
        float sc = scale[n], sf = shift[n];
        #pragma unroll
        for (int g2 = 0; g2 < 8; ++g2) {
            int m = m0 + mbase + g2;
            if (m < M) {
                float val = fmaxf(acc[j][g2] * sc + sf, 0.0f);
                if (mode == 0) out16[(size_t)m * cstride + coffset + n] = (_Float16)val;
                else           out32[(size_t)n * M + m] = val;
            }
        }
    }
}

// ======================================================================
// ConvTranspose with kernel==stride: per-tap 1x1 GEMM (K = Cin) with
// scattered stores into the upsampled NHWC concat buffer. blockIdx.z = tap.
// ======================================================================
__global__ __launch_bounds__(256) void k_deconv_wmma(
    const _Float16* __restrict__ in, int Win, int Cin,
    const _Float16* __restrict__ wt,
    const float* __restrict__ scale, const float* __restrict__ shift,
    _Float16* __restrict__ out, int M, int k, int Cout, int coffset, int cstride)
{
    __shared__ _Float16 sA[BM * LDT];
    __shared__ _Float16 sB[BN * LDT];

    const int m0 = blockIdx.x * BM;
    const int n0 = blockIdx.y * BN;
    const int tap = blockIdx.z;
    const int ky = tap / k, kx = tap % k;
    const int tid = threadIdx.x;
    const int lane = tid & 31, wv = tid >> 5;
    const int Wout = Win * k;

    const int ar = tid >> 1;
    const int ah = (tid & 1) * 16;
    const int am = m0 + ar;
    const bool valid = (am < M);
    const _Float16* const asrc = in + (size_t)am * Cin + ah;
    uint4* const adst = (uint4*)(sA + ar * LDT + ah);

    const int bn = tid >> 1;
    const int bh = (tid & 1) * 16;
    const bool bldr = (tid < 128);
    const _Float16* const bsrc = wt + ((size_t)tap * Cout + n0 + bn) * Cin + bh;
    uint4* const bdst = (uint4*)(sB + bn * LDT + bh);

    const int fr = lane & 15;
    const int kb = (lane >> 4) * 8;
    const _Float16* const paf = sA + (wv * 16 + fr) * LDT + kb;
    const _Float16* const pbf = sB + fr * LDT + kb;

    v8f acc[4] = {};

    uint4 a0 = {0,0,0,0}, a1 = {0,0,0,0};
    uint4 b0 = {0,0,0,0}, b1 = {0,0,0,0};
    if (valid) { const uint4* s = (const uint4*)asrc; a0 = s[0]; a1 = s[1]; }
    if (bldr)  { const uint4* s = (const uint4*)bsrc; b0 = s[0]; b1 = s[1]; }

    for (int c0 = 0; c0 < Cin; c0 += BK) {
        adst[0] = a0; adst[1] = a1;
        if (bldr) { bdst[0] = b0; bdst[1] = b1; }
        __syncthreads();

        if (c0 + BK < Cin) {
            a0 = uint4{0,0,0,0}; a1 = uint4{0,0,0,0};
            if (valid) { const uint4* s = (const uint4*)(asrc + c0 + BK); a0 = s[0]; a1 = s[1]; }
            if (bldr)  { const uint4* s = (const uint4*)(bsrc + c0 + BK); b0 = s[0]; b1 = s[1]; }
        }

        v8h alo = *(const v8h*)paf;
        v8h ahi = *(const v8h*)(paf + 16);
        v16h a = __builtin_shufflevector(alo, ahi, 0,1,2,3,4,5,6,7,8,9,10,11,12,13,14,15);

        #pragma unroll
        for (int j = 0; j < 4; ++j) {
            const _Float16* pb = pbf + j * 16 * LDT;
            v8h blo = *(const v8h*)pb;
            v8h bhi = *(const v8h*)(pb + 16);
            v16h bb = __builtin_shufflevector(blo, bhi, 0,1,2,3,4,5,6,7,8,9,10,11,12,13,14,15);
            acc[j] = __builtin_amdgcn_wmma_f32_16x16x32_f16(
                false, a, false, bb, (short)0, acc[j], false, false);
        }
        __syncthreads();
    }

    const int mbase = wv * 16 + (lane >> 4) * 8;
    const int nf = lane & 15;
    #pragma unroll
    for (int j = 0; j < 4; ++j) {
        int n = n0 + j * 16 + nf;
        float sc = scale[n], sf = shift[n];
        #pragma unroll
        for (int g2 = 0; g2 < 8; ++g2) {
            int m = m0 + mbase + g2;
            if (m < M) {
                int iy = m / Win, ix = m % Win;
                size_t opix = (size_t)(iy * k + ky) * Wout + (ix * k + kx);
                float val = fmaxf(acc[j][g2] * sc + sf, 0.0f);
                out[opix * cstride + coffset + n] = (_Float16)val;
            }
        }
    }
}

// ======================================================================
// host orchestration
// ======================================================================
extern "C" void kernel_launch(void* const* d_in, const int* in_sizes, int n_in,
                              void* d_out, int out_size, void* d_ws, size_t ws_size,
                              hipStream_t stream)
{
    (void)n_in; (void)out_size; (void)ws_size;
    const float* points = (const float*)d_in[0];
    const int npts = in_sizes[0] / 5;

    // layer index: 0=pfn, 1..13 backbone convs, 14=deblock conv, 15/16=deconvs, 17=final
    static const int cinL[18]  = {9, 64,64,64, 64,128,128,128,128, 128,256,256,256,256, 64,128,256, 384};
    static const int coutL[18] = {64,64,64,64, 128,128,128,128,128, 256,256,256,256,256, 128,128,128, 256};
    static const int ksL[18]   = {1, 3,3,3, 3,3,3,3,3, 3,3,3,3,3, 3, 2, 4, 3};
    static const bool decL[18] = {false, false,false,false, false,false,false,false,false,
                                  false,false,false,false,false, false, true, true, false};

    const float *gP[18], *bP[18], *mP[18], *vP[18], *wP[18];
    {
        int p = 1;  // params flatten: each layer dict is {g,b,m,v,w}
        for (int L = 0; L < 18; ++L) {
            gP[L] = (const float*)d_in[p+0];
            bP[L] = (const float*)d_in[p+1];
            mP[L] = (const float*)d_in[p+2];
            vP[L] = (const float*)d_in[p+3];
            wP[L] = (const float*)d_in[p+4];
            p += 5;
        }
    }

    // ---- workspace bump allocation ----
    char* base = (char*)d_ws; size_t off = 0;
    auto alloc = [&](size_t bytes) -> void* {
        void* r = base + off; off += (bytes + 255) & ~(size_t)255; return r;
    };
    int*   vids   = (int*)  alloc((size_t)npts * 4);
    float* counts = (float*)alloc((size_t)NSEG * 4);
    float* sums   = (float*)alloc((size_t)NSEG * 12);
    float* pooled = (float*)alloc((size_t)NSEG * 64 * 4);
    _Float16* P0 = (_Float16*)alloc((size_t)NSEG * 64 * 2);   // also holds 200^2 x 128
    _Float16* P1 = (_Float16*)alloc((size_t)NSEG * 64 * 2);
    _Float16* P2 = (_Float16*)alloc((size_t)NSEG * 64 * 2);
    _Float16* P3 = (_Float16*)alloc((size_t)100*100*256 * 2);
    _Float16* concat = (_Float16*)alloc((size_t)NSEG * 384 * 2);
    float* scl[18]; float* shf[18]; _Float16* wt[18];
    for (int L = 0; L < 18; ++L) {
        scl[L] = (float*)alloc((size_t)coutL[L] * 4);
        shf[L] = (float*)alloc((size_t)coutL[L] * 4);
        if (L == 0) { wt[L] = nullptr; continue; }
        size_t nh = decL[L] ? (size_t)ksL[L]*ksL[L]*coutL[L]*cinL[L]
                            : (size_t)coutL[L]*9*cinL[L];
        wt[L] = (_Float16*)alloc(nh * 2);
    }

    // ---- zero atomic targets ----
    hipMemsetAsync(counts, 0, (size_t)NSEG * 4, stream);
    hipMemsetAsync(sums,   0, (size_t)NSEG * 12, stream);
    hipMemsetAsync(pooled, 0, (size_t)NSEG * 64 * 4, stream);

    // ---- BN folding + weight repacks ----
    for (int L = 0; L < 18; ++L) {
        int n = coutL[L];
        k_bnfold<<<(n + 63) / 64, 64, 0, stream>>>(gP[L], bP[L], mP[L], vP[L], scl[L], shf[L], n);
        if (L == 0) continue;
        if (decL[L]) {
            int tot = cinL[L] * coutL[L] * ksL[L] * ksL[L];
            k_repack_deconv<<<(tot + 255) / 256, 256, 0, stream>>>(wP[L], wt[L], cinL[L], coutL[L], ksL[L]);
        } else {
            int tot = coutL[L] * cinL[L] * 9;
            k_repack_conv<<<(tot + 255) / 256, 256, 0, stream>>>(wP[L], wt[L], coutL[L], cinL[L]);
        }
    }

    // ---- voxelize + PFN + canvas ----
    k_voxelize<<<(npts + 255) / 256, 256, 0, stream>>>(points, npts, counts, sums, vids);
    k_pfn<<<(npts + 255) / 256, 256, 0, stream>>>(points, npts, vids, counts, sums,
                                                  wP[0], scl[0], shf[0], pooled);
    {
        int n = NSEG * 64;
        k_cvt_f32_f16<<<(n + 255) / 256, 256, 0, stream>>>(pooled, P0, n);
    }

    // ---- conv chain ----
    auto conv = [&](const _Float16* inp, _Float16* o16, float* o32,
                    int Hin, int Win, int L, int stride, int coff, int cstr, int mode) {
        int Hout = Hin / stride, Wout = Win / stride;
        int M = Hout * Wout;
        dim3 grid((M + BM - 1) / BM, coutL[L] / BN, 1);
        k_conv3x3_wmma<<<grid, 256, 0, stream>>>(inp, Hin, Win, cinL[L], stride,
                                                 wt[L], scl[L], shf[L], o16, o32,
                                                 Hout, Wout, coff, cstr, mode);
    };
    auto deconv = [&](const _Float16* inp, int Hin, int Win, int L, int coff, int cstr) {
        int M = Hin * Win;
        int k = ksL[L];
        dim3 grid((M + BM - 1) / BM, coutL[L] / BN, k * k);
        k_deconv_wmma<<<grid, 256, 0, stream>>>(inp, Win, cinL[L], wt[L], scl[L], shf[L],
                                                concat, M, k, coutL[L], coff, cstr);
    };

    // block 0 @ 400x400 (64ch): P0 -> P1 -> P2 -> P1   (feats0 = P1)
    conv(P0, P1, nullptr, 400, 400, 1, 1, 0, 64, 0);
    conv(P1, P2, nullptr, 400, 400, 2, 1, 0, 64, 0);
    conv(P2, P1, nullptr, 400, 400, 3, 1, 0, 64, 0);
    // block 1 -> 200x200 (128ch): P1 -> P0 -> P2 -> P0 -> P2 -> P0   (feats1 = P0)
    conv(P1, P0, nullptr, 400, 400, 4, 2, 0, 128, 0);
    conv(P0, P2, nullptr, 200, 200, 5, 1, 0, 128, 0);
    conv(P2, P0, nullptr, 200, 200, 6, 1, 0, 128, 0);
    conv(P0, P2, nullptr, 200, 200, 7, 1, 0, 128, 0);
    conv(P2, P0, nullptr, 200, 200, 8, 1, 0, 128, 0);
    // block 2 -> 100x100 (256ch): P0 -> P2 -> P3 -> P2 -> P3 -> P2   (feats2 = P2)
    conv(P0, P2, nullptr, 200, 200,  9, 2, 0, 256, 0);
    conv(P2, P3, nullptr, 100, 100, 10, 1, 0, 256, 0);
    conv(P3, P2, nullptr, 100, 100, 11, 1, 0, 256, 0);
    conv(P2, P3, nullptr, 100, 100, 12, 1, 0, 256, 0);
    conv(P3, P2, nullptr, 100, 100, 13, 1, 0, 256, 0);
    // deblocks -> concat (NHWC, 384 ch)
    conv(P1, concat, nullptr, 400, 400, 14, 1, 0, 384, 0);   // u0: ch [0,128)
    deconv(P0, 200, 200, 15, 128, 384);                      // u1: ch [128,256)
    deconv(P2, 100, 100, 16, 256, 384);                      // u2: ch [256,384)
    // final 3x3 conv 384->256, f32 NCHW straight to d_out
    conv(concat, nullptr, (float*)d_out, 400, 400, 17, 1, 0, 0, 1);
}
